// DecoderLSTM_19902878449727
// MI455X (gfx1250) — matrix operand order repaired and checked by reading, not compiled, not run on hardware
//
#include <hip/hip_runtime.h>
#include <math.h>

// ---------------- shapes ----------------
static constexpr int Bk = 16, Tk = 512, Vk = 4096, Hk = 1024;
static constexpr int G4H = 4 * Hk;      // 4096

// ---------------- types -----------------
typedef unsigned int u32;
typedef __attribute__((ext_vector_type(16))) __bf16 v16bf;
typedef __attribute__((ext_vector_type(8)))  float  v8f;
typedef __attribute__((ext_vector_type(4)))  u32    u32x4;

union FragU { u32x4 q[2]; v16bf v; };

__device__ __forceinline__ unsigned short f2bf(float f) {
    u32 bits = __float_as_uint(f);
    u32 r = bits + 0x7FFFu + ((bits >> 16) & 1u);   // round-to-nearest-even
    return (unsigned short)(r >> 16);
}

__device__ __forceinline__ v8f wmma_bf16(const FragU& a, const FragU& b, v8f c) {
    return __builtin_amdgcn_wmma_f32_16x16x32_bf16(
        false, a.v, false, b.v, (short)0, c, false, false);
}

// ------------- fp32 -> bf16 flat convert -------------
__global__ __launch_bounds__(256)
void cvt_f32_bf16(const float* __restrict__ in, unsigned short* __restrict__ out, long n) {
    long i = (long)blockIdx.x * blockDim.x + threadIdx.x;
    if (i < n) out[i] = f2bf(in[i]);
}

// ------------- fp32 [K,N] -> bf16 [N,K] transpose convert -------------
__global__ __launch_bounds__(256)
void cvt_transpose_bf16(const float* __restrict__ in, unsigned short* __restrict__ out,
                        int K, int N) {
    long i = (long)blockIdx.x * blockDim.x + threadIdx.x;
    long total = (long)K * N;
    if (i < total) {
        int k = (int)(i / N);
        int n = (int)(i % N);
        out[(long)n * K + k] = f2bf(in[i]);
    }
}

// ------------- zero fill (float4) -------------
__global__ __launch_bounds__(256)
void zero_f4(float4* __restrict__ p, long n4) {
    long i = (long)blockIdx.x * blockDim.x + threadIdx.x;
    long stride = (long)gridDim.x * blockDim.x;
    float4 z; z.x = z.y = z.z = z.w = 0.0f;
    for (; i < n4; i += stride) p[i] = z;
}

// ============================================================
// Big GEMM: C[M,N] = A[M,K](bf16,row) * Bt[N,K](bf16, pre-transposed)
// block = 4 waves; wave computes 32 rows x 64 cols (2x4 wmma tiles),
// K-loop software-pipelined (register double-buffer, 64-wide steps).
// Requires M % 128 == 0, N % 64 == 0, K % 64 == 0.
// ============================================================
__global__ __launch_bounds__(128)
void wmma_gemm_bf16(const unsigned short* __restrict__ A,
                    const unsigned short* __restrict__ Bt,
                    float* __restrict__ C,
                    int M, int N, int K) {
    const int lane = threadIdx.x & 31;
    const int wave = threadIdx.x >> 5;
    const int half = lane >> 4;       // 0: K-lo slice, 1: K-hi slice
    const int l15  = lane & 15;
    const int m0 = (blockIdx.y * 4 + wave) * 32;   // 2 M-tiles per wave
    const int n0 = blockIdx.x * 64;

    // 32-bit per-lane offsets -> saddr + voffset addressing
    u32 abase[2], bbase[4];
    abase[0] = (u32)(m0 + l15) * (u32)K + (u32)(half * 8);
    abase[1] = (u32)(m0 + 16 + l15) * (u32)K + (u32)(half * 8);
#pragma unroll
    for (int s = 0; s < 4; ++s)
        bbase[s] = (u32)(n0 + s * 16 + l15) * (u32)K + (u32)(half * 8);

    auto ldA = [&](FragU& f, int mi, int k0) {
        const unsigned short* p = A + abase[mi] + k0;
        f.q[0] = *(const u32x4*)(p);
        f.q[1] = *(const u32x4*)(p + 16);
    };
    auto ldB = [&](FragU& f, int s, int k0) {
        const unsigned short* p = Bt + bbase[s] + k0;
        f.q[0] = *(const u32x4*)(p);
        f.q[1] = *(const u32x4*)(p + 16);
    };

    v8f acc[2][4] = {};
    FragU a0[2], a1[2], b0[4], b1[4];

    // prologue: stage 0 at k=0
#pragma unroll
    for (int mi = 0; mi < 2; ++mi) ldA(a0[mi], mi, 0);
#pragma unroll
    for (int s = 0; s < 4; ++s)    ldB(b0[s], s, 0);

    for (int k0 = 0; k0 + 64 <= K; k0 += 64) {
        if (k0 + 1024 < K) __builtin_prefetch((const void*)(A + abase[0] + k0 + 1024), 0, 1);
        // issue stage 1 loads (k0+32) before consuming stage 0
#pragma unroll
        for (int mi = 0; mi < 2; ++mi) ldA(a1[mi], mi, k0 + 32);
#pragma unroll
        for (int s = 0; s < 4; ++s)    ldB(b1[s], s, k0 + 32);
#pragma unroll
        for (int mi = 0; mi < 2; ++mi)
#pragma unroll
            for (int s = 0; s < 4; ++s)
                acc[mi][s] = wmma_bf16(a0[mi], b0[s], acc[mi][s]);

        if (k0 + 64 < K) {          // refill stage 0 (k0+64) before consuming stage 1
#pragma unroll
            for (int mi = 0; mi < 2; ++mi) ldA(a0[mi], mi, k0 + 64);
#pragma unroll
            for (int s = 0; s < 4; ++s)    ldB(b0[s], s, k0 + 64);
        }
#pragma unroll
        for (int mi = 0; mi < 2; ++mi)
#pragma unroll
            for (int s = 0; s < 4; ++s)
                acc[mi][s] = wmma_bf16(a1[mi], b1[s], acc[mi][s]);
    }

#pragma unroll
    for (int mi = 0; mi < 2; ++mi)
#pragma unroll
        for (int s = 0; s < 4; ++s)
#pragma unroll
            for (int r = 0; r < 8; ++r) {
                u32 m = (u32)(m0 + mi * 16 + r + half * 8);
                u32 n = (u32)(n0 + s * 16 + l15);
                C[m * (u32)N + n] = acc[mi][s][r];
            }
}

// ============================================================
// LSTM step: one wave per 16 hidden columns. Computes all four
// gate tiles (i,f,g,o) = h @ Wh for cols [n0,n0+16) -> cell update
// is lane-local. Reads xpre/bias, updates c in place, writes h(bf16).
// ============================================================
__global__ __launch_bounds__(32)
void lstm_step(const unsigned short* __restrict__ hb,    // [16 x H] bf16
               const unsigned short* __restrict__ WhT,   // [4H x H] bf16
               const float* __restrict__ x_pre,          // [B*T x 4H] fp32
               const float* __restrict__ bias,           // [4H]
               float* __restrict__ c,                    // [16 x H] fp32 (in/out)
               unsigned short* __restrict__ hb_out,      // [16 x H] bf16
               int t) {
    const int lane = threadIdx.x & 31;
    const int half = lane >> 4;
    const int l15  = lane & 15;
    const int n0 = blockIdx.x * 16;

    const u32 abase = (u32)l15 * Hk + (u32)(half * 8);
    u32 bbase[4];
#pragma unroll
    for (int g = 0; g < 4; ++g)
        bbase[g] = (u32)(g * Hk + n0 + l15) * (u32)Hk + (u32)(half * 8);

    auto ldA = [&](FragU& f, int k0) {
        const unsigned short* p = hb + abase + k0;
        f.q[0] = *(const u32x4*)(p);
        f.q[1] = *(const u32x4*)(p + 16);
    };
    auto ldB = [&](FragU& f, int g, int k0) {
        const unsigned short* p = WhT + bbase[g] + k0;
        f.q[0] = *(const u32x4*)(p);
        f.q[1] = *(const u32x4*)(p + 16);
    };

    v8f acc[4] = {};
    FragU a0, a1, b0[4], b1[4];
    ldA(a0, 0);
#pragma unroll
    for (int g = 0; g < 4; ++g) ldB(b0[g], g, 0);

    for (int k0 = 0; k0 + 64 <= Hk; k0 += 64) {
        ldA(a1, k0 + 32);
#pragma unroll
        for (int g = 0; g < 4; ++g) ldB(b1[g], g, k0 + 32);
#pragma unroll
        for (int g = 0; g < 4; ++g) acc[g] = wmma_bf16(a0, b0[g], acc[g]);
        if (k0 + 64 < Hk) {
            ldA(a0, k0 + 64);
#pragma unroll
            for (int g = 0; g < 4; ++g) ldB(b0[g], g, k0 + 64);
        }
#pragma unroll
        for (int g = 0; g < 4; ++g) acc[g] = wmma_bf16(a1, b1[g], acc[g]);
    }

    const int n = n0 + l15;
#pragma unroll
    for (int r = 0; r < 8; ++r) {
        int m = r + half * 8;                                // batch index
        u32 xprow = ((u32)m * Tk + (u32)t) * (u32)G4H;
        float gi = acc[0][r] + x_pre[xprow + 0 * Hk + n] + bias[0 * Hk + n];
        float gf = acc[1][r] + x_pre[xprow + 1 * Hk + n] + bias[1 * Hk + n];
        float gg = acc[2][r] + x_pre[xprow + 2 * Hk + n] + bias[2 * Hk + n];
        float go = acc[3][r] + x_pre[xprow + 3 * Hk + n] + bias[3 * Hk + n];
        float si = 1.0f / (1.0f + __expf(-gi));
        float sf = 1.0f / (1.0f + __expf(-gf));
        float so = 1.0f / (1.0f + __expf(-go));
        float tg = tanhf(gg);
        u32 ci = (u32)m * Hk + (u32)n;
        float cn = sf * c[ci] + si * tg;
        c[ci] = cn;
        hb_out[ci] = f2bf(so * tanhf(cn));
    }
}

// ============================================================
// Logits: logits[:,t,:] = h @ Wd + bd. One wave per 64 vocab cols.
// ============================================================
__global__ __launch_bounds__(32)
void logits_step(const unsigned short* __restrict__ hb,   // [16 x H] bf16
                 const unsigned short* __restrict__ WdT,  // [V x H] bf16
                 const float* __restrict__ bd,            // [V]
                 float* __restrict__ out_logits,          // [B*T x V] fp32
                 int t) {
    const int lane = threadIdx.x & 31;
    const int half = lane >> 4;
    const int l15  = lane & 15;
    const int n0 = blockIdx.x * 64;

    const u32 abase = (u32)l15 * Hk + (u32)(half * 8);
    u32 bbase[4];
#pragma unroll
    for (int s = 0; s < 4; ++s)
        bbase[s] = (u32)(n0 + s * 16 + l15) * (u32)Hk + (u32)(half * 8);

    auto ldA = [&](FragU& f, int k0) {
        const unsigned short* p = hb + abase + k0;
        f.q[0] = *(const u32x4*)(p);
        f.q[1] = *(const u32x4*)(p + 16);
    };
    auto ldB = [&](FragU& f, int s, int k0) {
        const unsigned short* p = WdT + bbase[s] + k0;
        f.q[0] = *(const u32x4*)(p);
        f.q[1] = *(const u32x4*)(p + 16);
    };

    v8f acc[4] = {};
    FragU a0, a1, b0[4], b1[4];
    ldA(a0, 0);
#pragma unroll
    for (int s = 0; s < 4; ++s) ldB(b0[s], s, 0);

    for (int k0 = 0; k0 + 64 <= Hk; k0 += 64) {
        ldA(a1, k0 + 32);
#pragma unroll
        for (int s = 0; s < 4; ++s) ldB(b1[s], s, k0 + 32);
#pragma unroll
        for (int s = 0; s < 4; ++s) acc[s] = wmma_bf16(a0, b0[s], acc[s]);
        if (k0 + 64 < Hk) {
            ldA(a0, k0 + 64);
#pragma unroll
            for (int s = 0; s < 4; ++s) ldB(b0[s], s, k0 + 64);
        }
#pragma unroll
        for (int s = 0; s < 4; ++s) acc[s] = wmma_bf16(a1, b1[s], acc[s]);
    }

#pragma unroll
    for (int s = 0; s < 4; ++s)
#pragma unroll
        for (int r = 0; r < 8; ++r) {
            int m = r + half * 8;                  // batch
            int v = n0 + s * 16 + l15;             // vocab col
            out_logits[((u32)m * Tk + (u32)t) * (u32)Vk + (u32)v] = acc[s][r] + bd[v];
        }
}

// ============================================================
// Gumbel-argmax sampling: one wave per (b,t) row -> one-hot pred.
// ============================================================
__device__ __forceinline__ u32 hash_u32(u32 x) {
    x ^= x >> 16; x *= 0x7feb352dU;
    x ^= x >> 15; x *= 0x846ca68bU;
    x ^= x >> 16; return x;
}

__global__ __launch_bounds__(32)
void sample_kernel(const float* __restrict__ logits, float* __restrict__ preds) {
    const int row  = blockIdx.x;          // b*T + t
    const int lane = threadIdx.x;
    const float* lrow = logits + (size_t)row * Vk;
    float best = -1e30f;
    int   bi   = 0;
    for (int v = lane; v < Vk; v += 32) {
        u32 h = hash_u32((u32)row * 0x9E3779B9u + (u32)v * 0x85EBCA6Bu + 0x2545F491u);
        float u = (float)(h >> 8) * (1.0f / 16777216.0f) + 1e-12f;
        float gmb = -__logf(-__logf(u));
        float val = lrow[v] + gmb;
        if (val > best) { best = val; bi = v; }
    }
#pragma unroll
    for (int off = 16; off > 0; off >>= 1) {
        float ob = __shfl_xor(best, off, 32);
        int   oi = __shfl_xor(bi,   off, 32);
        if (ob > best || (ob == best && oi < bi)) { best = ob; bi = oi; }
    }
    if (lane == 0) preds[(size_t)row * Vk + bi] = 1.0f;
}

// ============================================================
// Host side
// ============================================================
extern "C" void kernel_launch(void* const* d_in, const int* in_sizes, int n_in,
                              void* d_out, int out_size, void* d_ws, size_t ws_size,
                              hipStream_t stream) {
    const float* x  = (const float*)d_in[0];   // [B,T,V]
    const float* c0 = (const float*)d_in[1];   // [B,H]
    const float* h0 = (const float*)d_in[2];   // [B,H]
    const float* Wi = (const float*)d_in[3];   // [V,4H]
    const float* Wh = (const float*)d_in[4];   // [H,4H]
    const float* b  = (const float*)d_in[5];   // [4H]
    const float* Wd = (const float*)d_in[6];   // [H,V]
    const float* bd = (const float*)d_in[7];   // [V]

    float* out_logits = (float*)d_out;                          // [B,T,V]
    float* out_preds  = out_logits + (size_t)Bk * Tk * Vk;      // [B,T,V]

    // ---- workspace carve-out ----
    char* ws = (char*)d_ws;
    const long nX = (long)Bk * Tk * Vk;            // 33.5M
    unsigned short* x_bf = (unsigned short*)ws;  ws += nX * 2;                  // 64 MB
    unsigned short* WiT  = (unsigned short*)ws;  ws += (long)G4H * Vk * 2;      // 32 MB  [4H x V]
    unsigned short* WhT  = (unsigned short*)ws;  ws += (long)G4H * Hk * 2;      //  8 MB  [4H x H]
    unsigned short* WdT  = (unsigned short*)ws;  ws += (long)Vk  * Hk * 2;      //  8 MB  [V x H]
    float*          x_pre = (float*)ws;          ws += (long)Bk * Tk * G4H * 4; // 128 MB [B*T x 4H]
    float*          c     = (float*)ws;          ws += (long)Bk * Hk * 4;       // 64 KB
    unsigned short* hb0   = (unsigned short*)ws; ws += (long)Bk * Hk * 2;       // 32 KB
    unsigned short* hb1   = (unsigned short*)ws; ws += (long)Bk * Hk * 2;       // 32 KB

    // ---- precision conversion / weight transposition ----
    cvt_f32_bf16<<<(unsigned)((nX + 255) / 256), 256, 0, stream>>>(x, x_bf, nX);
    {
        long n = (long)Vk * G4H;
        cvt_transpose_bf16<<<(unsigned)((n + 255) / 256), 256, 0, stream>>>(Wi, WiT, Vk, G4H);
    }
    {
        long n = (long)Hk * G4H;
        cvt_transpose_bf16<<<(unsigned)((n + 255) / 256), 256, 0, stream>>>(Wh, WhT, Hk, G4H);
    }
    {
        long n = (long)Hk * Vk;
        cvt_transpose_bf16<<<(unsigned)((n + 255) / 256), 256, 0, stream>>>(Wd, WdT, Hk, Vk);
    }
    {
        long n = (long)Bk * Hk;
        cvt_f32_bf16<<<(unsigned)((n + 255) / 256), 256, 0, stream>>>(h0, hb0, n);
    }
    hipMemcpyAsync(c, c0, (size_t)Bk * Hk * sizeof(float), hipMemcpyDeviceToDevice, stream);

    // ---- zero the one-hot preds output ----
    zero_f4<<<8192, 256, 0, stream>>>((float4*)out_preds, nX / 4);

    // ---- big input GEMM: x_pre[8192,4096] = x_bf @ Wi ----
    {
        dim3 grid(G4H / 64, (Bk * Tk) / 128);      // 64 x 64
        wmma_gemm_bf16<<<grid, 128, 0, stream>>>(x_bf, WiT, x_pre, Bk * Tk, G4H, Vk);
    }

    // ---- sequential scan over T ----
    for (int t = 0; t < Tk; ++t) {
        const unsigned short* hin = (t & 1) ? hb1 : hb0;
        unsigned short*      hout = (t & 1) ? hb0 : hb1;
        lstm_step<<<Hk / 16, 32, 0, stream>>>(hin, WhT, x_pre, b, c, hout, t);
        logits_step<<<Vk / 64, 32, 0, stream>>>(hout, WdT, bd, out_logits, t);
    }

    // ---- categorical (gumbel-argmax) sampling -> one-hot preds ----
    sample_kernel<<<Bk * Tk, 32, 0, stream>>>(out_logits, out_preds);
}
